// Eva_58497454571997
// MI455X (gfx1250) — compile-verified, matrix operand order checked by reading, not run on hardware
//
#include <hip/hip_runtime.h>
#include <hip/hip_bf16.h>

// ---------------------------------------------------------------------------
// EVA-ViT block (L=4, B=64, N=197, C=768, NH=12, HD=64, HF=2048) for gfx1250.
// Weight GEMMs: cooperative 64x256 block tiles (8 waves, 32x64 per wave),
// LDS double-buffered via GLOBAL_LOAD_ASYNC_TO_LDS_B128 (ASYNCcnt),
// v_wmma_f32_16x16x32_bf16 compute. Gate+X MLP GEMMs fused (N = 2*HF).
// Attention GEMMs: per-wave 16x64 strips, direct-global fragments.
// ---------------------------------------------------------------------------

typedef __bf16 bf16_t;
typedef __attribute__((ext_vector_type(16))) __bf16 v16bf;
typedef __attribute__((ext_vector_type(8)))  __bf16 v8bf;
typedef __attribute__((ext_vector_type(8)))  float  v8f;

#define Lc  4
#define Bc  64
#define Nc  197
#define Cc  768
#define NHc 12
#define HDc 64
#define HFc 2048
#define Tk  (Bc * Nc)        // 12608 token rows (multiple of 64)
#define NP  256              // padded token count (mult of 64; K mult of 32)

__device__ __forceinline__ bf16_t f2bf(float x) {
  unsigned u = __builtin_bit_cast(unsigned, x);
  u += 0x7FFFu + ((u >> 16) & 1u);             // round-to-nearest-even
  unsigned short h = (unsigned short)(u >> 16);
  return __builtin_bit_cast(bf16_t, h);
}
__device__ __forceinline__ float bf2f(bf16_t b) {
  unsigned short h = __builtin_bit_cast(unsigned short, b);
  unsigned u = ((unsigned)h) << 16;
  return __builtin_bit_cast(float, u);
}

// Load one 16x32 bf16 fragment half-row for this lane (global or LDS pointer).
// p = row_base + hi*8 ; lanes 0-15: K 0..7 then 16..23 ; lanes 16-31: +8.
__device__ __forceinline__ v16bf load_frag(const bf16_t* p) {
  v8bf lo = *(const v8bf*)p;
  v8bf hi = *(const v8bf*)(p + 16);
  v16bf f;
#pragma unroll
  for (int i = 0; i < 8; ++i) { f[i] = lo[i]; f[i + 8] = hi[i]; }
  return f;
}

enum { EP_BF16 = 0, EP_F32_BIAS = 1, EP_F32_RES = 2, EP_QKV = 3, EP_GX = 4 };

// ---------------------------------------------------------------------------
// Cooperative GEMM: out[m,n] = A[m,:].W[n,:] (+epilogue), A: MxK, W: NxK bf16.
// Block = 256 thr = 8 waves (2x4). Block tile 64(M)x256(N); wave tile 32x64.
// K consumed in 32-wide stages, double-buffered in LDS, filled by per-lane
// global_load_async_to_lds_b128 (5 loads/thread/stage) and drained with
// s_wait_asynccnt. Requires: M%64==0, N%256==0, K%32==0.
// ---------------------------------------------------------------------------
#define BM 64
#define BN 256
#define BK 32

template<int MODE>
__global__ __launch_bounds__(256) void gemm_coop(
    const bf16_t* __restrict__ A, int lda,
    const bf16_t* __restrict__ W, int ldw,
    float* __restrict__ Out, int ldo,
    const float* __restrict__ bias0, const float* __restrict__ bias1, int K)
{
  __shared__ alignas(16) bf16_t shA[2][BM * BK];   //  4 KB per buffer
  __shared__ alignas(16) bf16_t shB[2][BN * BK];   // 16 KB per buffer

  const int tid  = (int)threadIdx.x;
  const int lane = tid & 31;
  const int wave = tid >> 5;
  const int wm   = wave >> 2;            // 0..1  (M)
  const int wn   = wave & 3;             // 0..3  (N)
  const int r    = lane & 15;
  const int hi   = lane >> 4;
  const int bn   = (int)blockIdx.x;      // N / 256
  const int bm   = (int)blockIdx.y;      // M / 64

  const bf16_t* Ablk = A + (long long)bm * BM * lda;
  const bf16_t* Wblk = W + (long long)bn * BN * ldw;

  // Per-stage async fill: A = 64 rows x 4 16B-chunks (1/thread),
  //                       B = 256 rows x 4 16B-chunks (4/thread).
  const int rowA = tid >> 2, chA = (tid & 3) * 8;

  auto issue = [&](int buf, int k0) {
    {
      const bf16_t* src = Ablk + (long long)rowA * lda + k0 + chA;
      unsigned dst = (unsigned)(size_t)&shA[buf][rowA * BK + chA];
      unsigned long long ga = (unsigned long long)(size_t)src;
      asm volatile("global_load_async_to_lds_b128 %0, %1, off"
                   :: "v"(dst), "v"(ga) : "memory");
    }
#pragma unroll
    for (int i = 0; i < 4; ++i) {
      int c = tid + i * 256;             // 0..1023
      int rb = c >> 2, cb = (c & 3) * 8;
      const bf16_t* src = Wblk + (long long)rb * ldw + k0 + cb;
      unsigned dst = (unsigned)(size_t)&shB[buf][rb * BK + cb];
      unsigned long long ga = (unsigned long long)(size_t)src;
      asm volatile("global_load_async_to_lds_b128 %0, %1, off"
                   :: "v"(dst), "v"(ga) : "memory");
    }
  };

  v8f acc[2][4] = {{{}, {}, {}, {}}, {{}, {}, {}, {}}};

  const int nstages = K / BK;
  issue(0, 0);
  int buf = 0;
  for (int s = 0; s < nstages; ++s) {
    if (s + 1 < nstages) {
      issue(buf ^ 1, (s + 1) * BK);
      asm volatile("s_wait_asynccnt 0x5" ::: "memory");  // this stage's 5 done
    } else {
      asm volatile("s_wait_asynccnt 0x0" ::: "memory");
    }
    __syncthreads();                                     // tile visible to all waves

    const bf16_t* At = &shA[buf][(wm * 32 + r) * BK + hi * 8];
    const bf16_t* Bt = &shB[buf][(wn * 64 + r) * BK + hi * 8];
    v16bf a[2], b[4];
    a[0] = load_frag(At);
    a[1] = load_frag(At + 16 * BK);
#pragma unroll
    for (int fj = 0; fj < 4; ++fj)
      b[fj] = load_frag(Bt + fj * 16 * BK);
#pragma unroll
    for (int fi = 0; fi < 2; ++fi)
#pragma unroll
      for (int fj = 0; fj < 4; ++fj)
        acc[fi][fj] = __builtin_amdgcn_wmma_f32_16x16x32_bf16(
            false, a[fi], false, b[fj], (short)0, acc[fi][fj], false, false);

    __syncthreads();                                     // all reads done before refill
    buf ^= 1;
  }

  const int mbase = bm * BM + wm * 32;
  const int nbase = bn * BN + wn * 64;
#pragma unroll
  for (int fi = 0; fi < 2; ++fi) {
#pragma unroll
    for (int j = 0; j < 8; ++j) {
      const int m = mbase + fi * 16 + hi * 8 + j;
      const long long rowoff = (long long)m * ldo;
#pragma unroll
      for (int fj = 0; fj < 4; ++fj) {
        const int n = nbase + fj * 16 + r;
        const long long idx = rowoff + n;
        const float v = acc[fi][fj][j];
        if (MODE == EP_F32_BIAS) {
          Out[idx] = v + (bias0 ? bias0[n] : 0.f);
        } else if (MODE == EP_F32_RES) {
          Out[idx] = Out[idx] + v + (bias0 ? bias0[n] : 0.f);
        } else if (MODE == EP_QKV) { // [q_bias | 0 | v_bias]
          float bsv = (n < Cc) ? bias0[n] : ((n < 2 * Cc) ? 0.f : bias1[n - 2 * Cc]);
          Out[idx] = v + bsv;
        } else { // EP_GX: [b1g | b1x]
          float bsv = (n < HFc) ? bias0[n] : bias1[n - HFc];
          Out[idx] = v + bsv;
        }
      }
    }
  }
}

// ---------------------------------------------------------------------------
// Per-wave batched GEMM (direct-global fragments) for the attention matmuls.
// Wave computes a 16(M) x 64(N) strip; 1 A-frag + 4 B-frags -> 4 WMMA/Kstep.
// ---------------------------------------------------------------------------
template<int MODE>
__global__ __launch_bounds__(256) void gemm_nt(
    const bf16_t* __restrict__ A, long long sA, int lda,
    const bf16_t* __restrict__ W, long long sW, int ldw,
    void* __restrict__ Out, int nhDiv, long long ob0, long long ob1, int ldo,
    int Mtiles, int Nstrips, int K, int Mreal)
{
  const int wv = (int)(blockIdx.x * 8u + (threadIdx.x >> 5));
  if (wv >= Mtiles * Nstrips) return;          // wave-uniform: EXEC stays full
  const int lane = threadIdx.x & 31;
  const int z    = (int)blockIdx.y;
  const int mt   = wv / Nstrips;
  const int ns   = wv % Nstrips;
  const int r    = lane & 15;
  const int hi   = lane >> 4;

  const bf16_t* Ap = A + (long long)z * sA + ((long long)mt * 16 + r) * lda + hi * 8;
  const bf16_t* Wp = W + (long long)z * sW + ((long long)ns * 64 + r) * ldw + hi * 8;

  v8f acc0 = {}, acc1 = {}, acc2 = {}, acc3 = {};
  for (int k0 = 0; k0 < K; k0 += 32) {
    v16bf a  = load_frag(Ap + k0);
    v16bf b0 = load_frag(Wp + k0);
    v16bf b1 = load_frag(Wp + (long long)16 * ldw + k0);
    v16bf b2 = load_frag(Wp + (long long)32 * ldw + k0);
    v16bf b3 = load_frag(Wp + (long long)48 * ldw + k0);
    acc0 = __builtin_amdgcn_wmma_f32_16x16x32_bf16(false, a, false, b0, (short)0, acc0, false, false);
    acc1 = __builtin_amdgcn_wmma_f32_16x16x32_bf16(false, a, false, b1, (short)0, acc1, false, false);
    acc2 = __builtin_amdgcn_wmma_f32_16x16x32_bf16(false, a, false, b2, (short)0, acc2, false, false);
    acc3 = __builtin_amdgcn_wmma_f32_16x16x32_bf16(false, a, false, b3, (short)0, acc3, false, false);
  }

  const long long zb = z / nhDiv;
  const int zh = z % nhDiv;
  bf16_t* ob = (bf16_t*)Out;
  const long long obase = zb * ob0 + (long long)zh * ob1;
  const int ncol = ns * 64 + r;
#pragma unroll
  for (int j = 0; j < 8; ++j) {
    const int m = mt * 16 + hi * 8 + j;
    if (m < Mreal) {
      const long long rowoff = obase + (long long)m * ldo;
      float vals[4] = { acc0[j], acc1[j], acc2[j], acc3[j] };
#pragma unroll
      for (int s = 0; s < 4; ++s) {
        const long long idx = rowoff + ncol + s * 16;
        ob[idx] = f2bf(vals[s]);            // attention GEMMs are bf16-out
      }
    }
  }
}

// LayerNorm over last dim D, fp32 in -> bf16 out.
__global__ __launch_bounds__(256) void ln_kernel(
    const float* __restrict__ src, const float* __restrict__ g,
    const float* __restrict__ b, bf16_t* __restrict__ dst, int D)
{
  const int row = (int)blockIdx.x;
  const float* x = src + (long long)row * D;
  float s = 0.f, s2 = 0.f;
  for (int i = threadIdx.x; i < D; i += 256) { float v = x[i]; s += v; s2 += v * v; }
  __shared__ float r1[256], r2[256];
  r1[threadIdx.x] = s; r2[threadIdx.x] = s2;
  __syncthreads();
  for (int off = 128; off > 0; off >>= 1) {
    if ((int)threadIdx.x < off) {
      r1[threadIdx.x] += r1[threadIdx.x + off];
      r2[threadIdx.x] += r2[threadIdx.x + off];
    }
    __syncthreads();
  }
  const float mean = r1[0] / D;
  const float var  = r2[0] / D - mean * mean;
  const float inv  = rsqrtf(var + 1e-6f);
  bf16_t* o = dst + (long long)row * D;
  for (int i = threadIdx.x; i < D; i += 256)
    o[i] = f2bf((x[i] - mean) * inv * g[i] + b[i]);
}

// RoPE (interleaved rotate-half, skips CLS) + pack q*SCALE, k, v^T, padded.
__global__ __launch_bounds__(64) void rope_pack(
    const float* __restrict__ qkv, const float* __restrict__ rope,
    bf16_t* __restrict__ qb, bf16_t* __restrict__ kb, bf16_t* __restrict__ vt)
{
  const int d = (int)threadIdx.x;      // 0..63
  const int t = (int)blockIdx.x;       // 0..NP-1
  const int z = (int)blockIdx.y;       // 0..B*NH-1
  const int bb = z / NHc, hh = z % NHc;
  const long long qidx = (long long)z * (NP * HDc) + (long long)t * HDc + d;
  const long long vidx = (long long)z * (HDc * NP) + (long long)d * NP + t;
  if (t >= Nc) {
    bf16_t zr = f2bf(0.f);
    qb[qidx] = zr; kb[qidx] = zr; vt[vidx] = zr;
    return;
  }
  const long long base = ((long long)(bb * Nc + t)) * (3 * Cc) + hh * HDc;
  float q = qkv[base + d];
  float k = qkv[base + Cc + d];
  float v = qkv[base + 2 * Cc + d];
  if (t > 0) {
    const float* rp = rope + (long long)(t - 1) * (2 * HDc);
    float sn = rp[d], cs = rp[HDc + d];
    float qo = qkv[base + (d ^ 1)];
    float ko = qkv[base + Cc + (d ^ 1)];
    float qr = (d & 1) ? qo : -qo;     // rot: (-x1, x0, -x3, x2, ...)
    float kr = (d & 1) ? ko : -ko;
    q = q * cs + qr * sn;
    k = k * cs + kr * sn;
  }
  qb[qidx] = f2bf(q * 0.125f);         // SCALE = 64^-0.5
  kb[qidx] = f2bf(k);
  vt[vidx] = f2bf(v);
}

// Row softmax over first Nc columns of padded NPxNP score tiles.
__global__ __launch_bounds__(256) void softmax_kernel(
    const bf16_t* __restrict__ S, bf16_t* __restrict__ P)
{
  const int r = (int)blockIdx.x;
  const int z = (int)blockIdx.y;
  const int c = (int)threadIdx.x;
  const long long row = ((long long)z * NP + r) * NP;
  if (r >= Nc) { P[row + c] = f2bf(0.f); return; }
  float v = (c < Nc) ? bf2f(S[row + c]) : -3.0e38f;
  __shared__ float red[256];
  red[c] = v;
  __syncthreads();
  for (int off = 128; off > 0; off >>= 1) {
    if (c < off) red[c] = fmaxf(red[c], red[c + off]);
    __syncthreads();
  }
  const float mx = red[0];
  __syncthreads();
  float e = (c < Nc) ? __expf(v - mx) : 0.f;
  red[c] = e;
  __syncthreads();
  for (int off = 128; off > 0; off >>= 1) {
    if (c < off) red[c] += red[c + off];
    __syncthreads();
  }
  const float inv = 1.f / red[0];
  P[row + c] = f2bf(e * inv);
}

__global__ void cvt_bf16(const float* __restrict__ src, bf16_t* __restrict__ dst, long long n) {
  long long i = (long long)blockIdx.x * blockDim.x + threadIdx.x;
  if (i < n) dst[i] = f2bf(src[i]);
}

// u[row, col] = silu(gx[row, col]) * gx[row, HF + col]  over [Tk, HF]
__global__ void silu_mul2(const float* __restrict__ gx, float* __restrict__ u, long long n) {
  long long i = (long long)blockIdx.x * blockDim.x + threadIdx.x;
  if (i < n) {
    long long row = i >> 11;                  // HFc == 2048
    int col = (int)(i & (HFc - 1));
    float a = gx[row * (2 * HFc) + col];
    float b = gx[row * (2 * HFc) + HFc + col];
    u[i] = a / (1.f + __expf(-a)) * b;
  }
}

// ---------------------------------------------------------------------------

extern "C" void kernel_launch(void* const* d_in, const int* in_sizes, int n_in,
                              void* d_out, int out_size, void* d_ws, size_t ws_size,
                              hipStream_t stream) {
  (void)in_sizes; (void)n_in; (void)out_size; (void)ws_size;

  const float* x_in   = (const float*)d_in[0];
  const float* rope   = (const float*)d_in[1];
  const float* qkv_w  = (const float*)d_in[2];
  const float* q_bias = (const float*)d_in[3];
  const float* v_bias = (const float*)d_in[4];
  const float* proj_w = (const float*)d_in[5];
  const float* proj_b = (const float*)d_in[6];
  const float* n1_g   = (const float*)d_in[7];
  const float* n1_b   = (const float*)d_in[8];
  const float* n2_g   = (const float*)d_in[9];
  const float* n2_b   = (const float*)d_in[10];
  const float* w1g_f  = (const float*)d_in[11];
  const float* b1g    = (const float*)d_in[12];
  const float* w1x_f  = (const float*)d_in[13];
  const float* b1x    = (const float*)d_in[14];
  const float* nm_g   = (const float*)d_in[15];
  const float* nm_b   = (const float*)d_in[16];
  const float* w2_f   = (const float*)d_in[17];
  const float* b2     = (const float*)d_in[18];

  float* h = (float*)d_out;                       // residual stream, fp32 [Tk, C]

  char* ws = (char*)d_ws;
  size_t off = 0;
  auto alloc = [&](size_t bytes) -> char* {
    char* p = ws + off;
    off = (off + bytes + 255) & ~(size_t)255;
    return p;
  };
  const long long ZH = (long long)Bc * NHc;       // 768 attention batches
  bf16_t* wqkv  = (bf16_t*)alloc((size_t)Lc * 3 * Cc * Cc * 2);
  bf16_t* wproj = (bf16_t*)alloc((size_t)Lc * Cc * Cc * 2);
  bf16_t* w1gx  = (bf16_t*)alloc((size_t)Lc * 2 * HFc * Cc * 2);  // [b1g rows | b1x rows]
  bf16_t* w2w   = (bf16_t*)alloc((size_t)Lc * Cc * HFc * 2);
  bf16_t* ybf   = (bf16_t*)alloc((size_t)Tk * HFc * 2);
  float*  qkvf  = (float*) alloc((size_t)Tk * 3 * Cc * 4);
  bf16_t* qb    = (bf16_t*)alloc((size_t)ZH * NP * HDc * 2);
  bf16_t* kb    = (bf16_t*)alloc((size_t)ZH * NP * HDc * 2);
  bf16_t* vt    = (bf16_t*)alloc((size_t)ZH * HDc * NP * 2);
  bf16_t* Sb    = (bf16_t*)alloc((size_t)ZH * NP * NP * 2);
  bf16_t* Pb    = (bf16_t*)alloc((size_t)ZH * NP * NP * 2);
  bf16_t* Obf   = (bf16_t*)alloc((size_t)Tk * Cc * 2);
  float*  gxf   = (float*) alloc((size_t)Tk * 2 * HFc * 4);       // fused gate|x
  float*  uf    = (float*) alloc((size_t)Tk * HFc * 4);           // silu(g)*x

  hipMemcpyAsync(h, x_in, (size_t)Tk * Cc * sizeof(float),
                 hipMemcpyDeviceToDevice, stream);

  auto cvt = [&](const float* s, bf16_t* d, long long n) {
    unsigned blocks = (unsigned)((n + 255) / 256);
    cvt_bf16<<<blocks, 256, 0, stream>>>(s, d, n);
  };
  cvt(qkv_w,  wqkv,  (long long)Lc * 3 * Cc * Cc);
  cvt(proj_w, wproj, (long long)Lc * Cc * Cc);
  cvt(w2_f,   w2w,   (long long)Lc * Cc * HFc);
  for (int l = 0; l < Lc; ++l) {                  // pack w1g | w1x per layer
    cvt(w1g_f + (long long)l * HFc * Cc,
        w1gx + (long long)l * 2 * HFc * Cc,                 (long long)HFc * Cc);
    cvt(w1x_f + (long long)l * HFc * Cc,
        w1gx + (long long)l * 2 * HFc * Cc + (long long)HFc * Cc, (long long)HFc * Cc);
  }

  const unsigned GM = Tk / BM;                    // 197 row blocks

  for (int l = 0; l < Lc; ++l) {
    const bf16_t* Wqkv = wqkv  + (long long)l * 3 * Cc * Cc;
    const bf16_t* Wprj = wproj + (long long)l * Cc * Cc;
    const bf16_t* Wgx  = w1gx  + (long long)l * 2 * HFc * Cc;
    const bf16_t* W2   = w2w   + (long long)l * Cc * HFc;

    // ---- LN1 -> ybf
    ln_kernel<<<Tk, 256, 0, stream>>>(h, n1_g + (long long)l * Cc,
                                      n1_b + (long long)l * Cc, ybf, Cc);

    // ---- QKV = ybf @ Wqkv^T + [qb|0|vb]  (fp32, coop/async)
    gemm_coop<EP_QKV><<<dim3((3 * Cc) / BN, GM), 256, 0, stream>>>(
        ybf, Cc, Wqkv, Cc, qkvf, 3 * Cc,
        q_bias + (long long)l * Cc, v_bias + (long long)l * Cc, Cc);

    // ---- RoPE + pack q*scale, k, v^T (padded to NP)
    rope_pack<<<dim3(NP, (unsigned)ZH), 64, 0, stream>>>(qkvf, rope, qb, kb, vt);

    // ---- S = q @ k^T  (bf16, batched per head)
    {
      int Mt = NP / 16, Ns = NP / 64;
      dim3 grid((unsigned)((Mt * Ns + 7) / 8), (unsigned)ZH);
      gemm_nt<EP_BF16><<<grid, 256, 0, stream>>>(
          qb, (long long)NP * HDc, HDc, kb, (long long)NP * HDc, HDc,
          Sb, 1, (long long)NP * NP, 0LL, NP, Mt, Ns, HDc, NP);
    }

    // ---- softmax rows -> Pb
    softmax_kernel<<<dim3(NP, (unsigned)ZH), 256, 0, stream>>>(Sb, Pb);

    // ---- O = P @ V (A = P, W = V^T), scattered into [Tk, C] bf16
    {
      int Mt = NP / 16, Ns = HDc / 64;
      dim3 grid((unsigned)((Mt * Ns + 7) / 8), (unsigned)ZH);
      gemm_nt<EP_BF16><<<grid, 256, 0, stream>>>(
          Pb, (long long)NP * NP, NP, vt, (long long)HDc * NP, NP,
          Obf, NHc, (long long)Nc * Cc, (long long)HDc, Cc, Mt, Ns, NP, Nc);
    }

    // ---- h += O @ Wproj^T + proj_b  (coop/async)
    gemm_coop<EP_F32_RES><<<dim3(Cc / BN, GM), 256, 0, stream>>>(
        Obf, Cc, Wprj, Cc, h, Cc, proj_b + (long long)l * Cc, nullptr, Cc);

    // ---- LN2 -> ybf
    ln_kernel<<<Tk, 256, 0, stream>>>(h, n2_g + (long long)l * Cc,
                                      n2_b + (long long)l * Cc, ybf, Cc);

    // ---- fused [gate|x] GEMM (coop/async, N = 2*HF)
    gemm_coop<EP_GX><<<dim3((2 * HFc) / BN, GM), 256, 0, stream>>>(
        ybf, Cc, Wgx, Cc, gxf, 2 * HFc,
        b1g + (long long)l * HFc, b1x + (long long)l * HFc, Cc);

    // ---- u = silu(g) * x
    {
      long long n = (long long)Tk * HFc;
      silu_mul2<<<(unsigned)((n + 255) / 256), 256, 0, stream>>>(gxf, uf, n);
    }

    // ---- mid-LN over HF -> ybf (bf16)
    ln_kernel<<<Tk, 256, 0, stream>>>(uf, nm_g + (long long)l * HFc,
                                      nm_b + (long long)l * HFc, ybf, HFc);

    // ---- h += u @ W2^T + b2  (coop/async, K = HF)
    gemm_coop<EP_F32_RES><<<dim3(Cc / BN, GM), 256, 0, stream>>>(
        ybf, HFc, W2, HFc, h, Cc, b2 + (long long)l * Cc, nullptr, HFc);
  }
}
